// StructureModule_12867722018818
// MI455X (gfx1250) — matrix operand order repaired and checked by reading, not compiled, not run on hardware
//
#include <hip/hip_runtime.h>
#include <hip/hip_bf16.h>

// Problem constants
#define NRES 512
#define CS   384
#define CZ   128
#define CH   16
#define NH   12
#define NL   8
#define NREC 2

typedef __attribute__((ext_vector_type(16))) __bf16 v16bf;
typedef __attribute__((ext_vector_type(8)))  __bf16 v8bf;
typedef __attribute__((ext_vector_type(4)))  __bf16 v4bf;
typedef __attribute__((ext_vector_type(8)))  float  v8f;

__device__ __forceinline__ __bf16 cvtb(float v)  { return (__bf16)v; }
__device__ __forceinline__ __bf16 cvtb(__bf16 v) { return v; }

// raw 4-element vector type per source precision
template <typename T> struct Raw4;
template <> struct Raw4<float>  { using type = float4; };
template <> struct Raw4<__bf16> { using type = v4bf;  };

__device__ __forceinline__ v4bf cvt4(float4 r) {
    v4bf o;
    o[0] = (__bf16)r.x; o[1] = (__bf16)r.y; o[2] = (__bf16)r.z; o[3] = (__bf16)r.w;
    return o;
}
__device__ __forceinline__ v4bf cvt4(v4bf r) { return r; }

// ---------------------------------------------------------------------------
// Tiled WMMA GEMM:  C[M,N] = act(A[M,K] @ B[K,N] + bias[N])
//  - A row-major (f32/bf16), B row-major (f32/bf16); bf16 in LDS.
//  - B staged TRANSPOSED in LDS so both fragments are contiguous b128 reads.
//  - Ping-pong LDS, register prefetch of next tile, ONE barrier per stage.
//  - 8 waves/block (WM x WN); each wave owns TM x TN 16x16 tiles.
//  - BK = 64 -> 2*TM*TN WMMAs per stage.
//  - GM/GN: compile-time bounds guards (branch-free clamp + select).
//  Requires: K % BK == 0, K >= 2*BK; staging counts divisible by 256.
// ---------------------------------------------------------------------------
template <int BM, int BN, int BK, int WM, int WN, bool GM, bool GN,
          typename TA, typename TB>
__global__ __launch_bounds__(256) void gemm_wmma(
    const TA* __restrict__ A, int lda, long long sA,
    const TB* __restrict__ B, int ldb, long long sB,
    float* __restrict__ C, int ldc, long long sC,
    const float* __restrict__ bias,
    int M, int N, int K, int doRelu)
{
    static_assert(WM * WN == 8, "8 waves");
    constexpr int TM = BM / (16 * WM);
    constexpr int TN = BN / (16 * WN);
    constexpr int LP  = 8;                 // LDS row pad (bf16) = 16B
    constexpr int KQ  = BK / 4;
    constexpr int AIT = (BM * KQ) / 256;   // A staging iters (exact)
    constexpr int BIT = (BN * KQ) / 256;   // B staging iters (exact)
    static_assert((BM * KQ) % 256 == 0 && (BN * KQ) % 256 == 0, "exact staging");

    using RawA = typename Raw4<TA>::type;

    __shared__ __bf16 As[2][BM][BK + LP];  // row-major  [m][k]
    __shared__ __bf16 Bt[2][BN][BK + LP];  // transposed [n][k]

    const int tid  = threadIdx.x;
    const int lane = tid & 31;             // wave32
    const int wave = tid >> 5;
    const int wm = wave / WN;
    const int wn = wave % WN;
    const int m0 = blockIdx.y * BM;
    const int n0 = blockIdx.x * BN;

    const TA* Ab = A + (long long)blockIdx.z * sA;
    const TB* Bb = B + (long long)blockIdx.z * sB;
    float*    Cb = C + (long long)blockIdx.z * sC;

    // prefetch registers
    RawA ra[AIT];
    TB   rb[BIT][4];

    auto prefA = [&](int k0) {
#pragma unroll
        for (int it = 0; it < AIT; ++it) {
            int idx = tid + it * 256;
            int m = idx / KQ, kq = idx % KQ;
            int gm = m0 + m;
            int gmc = GM ? (gm < M ? gm : M - 1) : gm;
            ra[it] = *(const RawA*)(Ab + (long long)gmc * lda + (k0 + kq * 4));
        }
    };
    auto commitA = [&](int buf) {
#pragma unroll
        for (int it = 0; it < AIT; ++it) {
            int idx = tid + it * 256;
            int m = idx / KQ, kq = idx % KQ;
            v4bf d = cvt4(ra[it]);
            if (GM && (m0 + m) >= M) { d[0] = d[1] = d[2] = d[3] = (__bf16)0.f; }
            *(v4bf*)&As[buf][m][kq * 4] = d;
        }
    };
    auto prefB = [&](int k0) {
#pragma unroll
        for (int it = 0; it < BIT; ++it) {
            int idx = tid + it * 256;
            int n = idx % BN, kq = idx / BN;
            int gn = n0 + n;
            int gnc = GN ? (gn < N ? gn : N - 1) : gn;
            const TB* src = Bb + (long long)(k0 + kq * 4) * ldb + gnc;
            rb[it][0] = src[0];
            rb[it][1] = src[ldb];
            rb[it][2] = src[2 * ldb];
            rb[it][3] = src[3 * ldb];
        }
    };
    auto commitB = [&](int buf) {
#pragma unroll
        for (int it = 0; it < BIT; ++it) {
            int idx = tid + it * 256;
            int n = idx % BN, kq = idx / BN;
            v4bf d;
            d[0] = cvtb(rb[it][0]); d[1] = cvtb(rb[it][1]);
            d[2] = cvtb(rb[it][2]); d[3] = cvtb(rb[it][3]);
            if (GN && (n0 + n) >= N) { d[0] = d[1] = d[2] = d[3] = (__bf16)0.f; }
            *(v4bf*)&Bt[buf][n][kq * 4] = d;
        }
    };

    const v8f zero8 = {0.f, 0.f, 0.f, 0.f, 0.f, 0.f, 0.f, 0.f};
    v8f acc[TM][TN];
#pragma unroll
    for (int a = 0; a < TM; ++a)
#pragma unroll
        for (int b = 0; b < TN; ++b) acc[a][b] = zero8;

    const int mr = lane & 15;
    const int nr = lane & 15;
    const int ka = (lane < 16) ? 0 : 16;   // A frag: K 0..15 | 16..31
    const int kb = (lane < 16) ? 0 : 16;   // B frag: K 0..15 | 16..31

    // prologue: stage first tile
    prefA(0); prefB(0);
    commitA(0); commitB(0);
    __syncthreads();

    int buf = 0;
    for (int k0 = 0; k0 < K; k0 += BK) {
        const bool hasNext = (k0 + BK) < K;
        if (hasNext) { prefA(k0 + BK); prefB(k0 + BK); }   // overlap with WMMAs

        const __bf16(*Asb)[BK + LP] = As[buf];
        const __bf16(*Btb)[BK + LP] = Bt[buf];
#pragma unroll
        for (int kk = 0; kk < BK / 32; ++kk) {
            v16bf fa[TM], fb[TN];
#pragma unroll
            for (int a = 0; a < TM; ++a) {
                const __bf16* p = &Asb[(wm * TM + a) * 16 + mr][kk * 32 + ka];
                v8bf lo = *(const v8bf*)(p);
                v8bf hi = *(const v8bf*)(p + 8);
                fa[a] = __builtin_shufflevector(lo, hi, 0, 1, 2, 3, 4, 5, 6, 7,
                                                8, 9, 10, 11, 12, 13, 14, 15);
            }
#pragma unroll
            for (int b = 0; b < TN; ++b) {
                const __bf16* p = &Btb[(wn * TN + b) * 16 + nr][kk * 32 + kb];
                v8bf lo = *(const v8bf*)(p);
                v8bf hi = *(const v8bf*)(p + 8);
                fb[b] = __builtin_shufflevector(lo, hi, 0, 1, 2, 3, 4, 5, 6, 7,
                                                8, 9, 10, 11, 12, 13, 14, 15);
            }
#pragma unroll
            for (int a = 0; a < TM; ++a)
#pragma unroll
                for (int b = 0; b < TN; ++b)
                    acc[a][b] = __builtin_amdgcn_wmma_f32_16x16x32_bf16(
                        false, fa[a], false, fb[b], (short)0, acc[a][b],
                        false, false);
        }

        if (hasNext) { commitA(buf ^ 1); commitB(buf ^ 1); }
        __syncthreads();
        buf ^= 1;
    }

    // store: VGPR r -> M = r (lanes 0-15) / 8+r (lanes 16-31)
    const int mh = (lane < 16) ? 0 : 8;
#pragma unroll
    for (int a = 0; a < TM; ++a) {
#pragma unroll
        for (int b = 0; b < TN; ++b) {
            int gmb = m0 + (wm * TM + a) * 16 + mh;
            int gn  = n0 + (wn * TN + b) * 16 + nr;
            if (GN && gn >= N) continue;
            float bv = bias ? bias[gn] : 0.f;
#pragma unroll
            for (int r = 0; r < 8; ++r) {
                int gm = gmb + r;
                if (GM && gm >= M) continue;
                float o = acc[a][b][r] + bv;
                if (doRelu) o = fmaxf(o, 0.f);
                Cb[(long long)gm * ldc + gn] = o;
            }
        }
    }
}

// ---------------------------------------------------------------------------
// f32 -> bf16 bulk convert (one-time z conversion per call)
// ---------------------------------------------------------------------------
__global__ void cvt_bf16_kernel(const float* __restrict__ in,
                                __bf16* __restrict__ out, int n)
{
    int i = blockIdx.x * 256 + threadIdx.x;
    if (i < n) out[i] = (__bf16)in[i];
}

// init: s_buf = s_in, coords (d_out) = 0
__global__ void init_kernel(const float* __restrict__ s_in,
                            float* __restrict__ s_buf,
                            float* __restrict__ coords)
{
    int i = blockIdx.x * 256 + threadIdx.x;
    if (i < NRES * CS) s_buf[i] = s_in[i];
    if (i < NRES * 3)  coords[i] = 0.f;
}

// ---------------------------------------------------------------------------
// Fused attention scores + pair bias + distance bias + softmax over j.
// ---------------------------------------------------------------------------
__global__ __launch_bounds__(256) void attn_softmax_kernel(
    const float* __restrict__ q, const float* __restrict__ k,
    const float* __restrict__ bz, const float* __restrict__ coords,
    float* __restrict__ wgt)
{
    const int i = blockIdx.x;
    const int h = blockIdx.y;
    const int t = threadIdx.x;
    __shared__ float qs[CH];
    __shared__ float ci[3];
    __shared__ float red[256];

    if (t < CH) qs[t] = q[i * (NH * CH) + h * CH + t];
    if (t < 3)  ci[t] = coords[i * 3 + t];
    __syncthreads();

    float vals[2];
#pragma unroll
    for (int p = 0; p < 2; ++p) {
        int j = t + p * 256;
        const float4* kr = (const float4*)(k + j * (NH * CH) + h * CH);
        float dot = 0.f;
#pragma unroll
        for (int c4 = 0; c4 < CH / 4; ++c4) {
            float4 kv = kr[c4];
            dot += qs[c4 * 4 + 0] * kv.x + qs[c4 * 4 + 1] * kv.y +
                   qs[c4 * 4 + 2] * kv.z + qs[c4 * 4 + 3] * kv.w;
        }
        dot *= 0.25f;  // 1/sqrt(16)
        float bzv = bz[(long long)(i * NRES + j) * 16 + h];
        float dx = ci[0] - coords[j * 3 + 0];
        float dy = ci[1] - coords[j * 3 + 1];
        float dz = ci[2] - coords[j * 3 + 2];
        float d2 = dx * dx + dy * dy + dz * dz;
        float ms = (d2 <= 25.f) ? 1.f : ((d2 <= 225.f) ? 0.3f : 0.05f);
        vals[p] = dot + bzv + ms;
    }

    red[t] = fmaxf(vals[0], vals[1]);
    __syncthreads();
    for (int s = 128; s > 0; s >>= 1) {
        if (t < s) red[t] = fmaxf(red[t], red[t + s]);
        __syncthreads();
    }
    float mx = red[0];
    __syncthreads();

    float e0 = __expf(vals[0] - mx);
    float e1 = __expf(vals[1] - mx);
    red[t] = e0 + e1;
    __syncthreads();
    for (int s = 128; s > 0; s >>= 1) {
        if (t < s) red[t] += red[t + s];
        __syncthreads();
    }
    float inv = 1.f / red[0];

    float* wr = wgt + ((long long)i * NH + h) * NRES;
    wr[t]       = e0 * inv;
    wr[t + 256] = e1 * inv;
}

// ---------------------------------------------------------------------------
// scalar[i,h,c] = sum_j wgt[i,h,j] * v[j,h,c] -> obuf[i][h*144 + c]
// ---------------------------------------------------------------------------
__global__ __launch_bounds__(64) void attn_scalar_kernel(
    const float* __restrict__ wgt, const float* __restrict__ v,
    float* __restrict__ obuf)
{
    const int i = blockIdx.x;
    const int h = blockIdx.y;
    const int t = threadIdx.x;
    const int c = t & 15;
    const int p = t >> 4;
    const float* w = wgt + ((long long)i * NH + h) * NRES;
    float acc = 0.f;
    for (int j = p; j < NRES; j += 4)
        acc += w[j] * v[j * (NH * CH) + h * CH + c];
    __shared__ float red[64];
    red[t] = acc;
    __syncthreads();
    if (t < 32) red[t] += red[t + 32];
    __syncthreads();
    if (t < 16)
        obuf[(long long)i * (NH * (CH + CZ)) + h * (CH + CZ) + c] = red[t] + red[t + 16];
}

// ---------------------------------------------------------------------------
// s = LayerNorm(s + x) * g + b
// ---------------------------------------------------------------------------
__global__ __launch_bounds__(128) void add_ln_kernel(
    float* __restrict__ s, const float* __restrict__ x,
    const float* __restrict__ g, const float* __restrict__ b)
{
    const int i = blockIdx.x;
    const int t = threadIdx.x;
    __shared__ float red[128];
    float vals[3];
    float sum = 0.f;
#pragma unroll
    for (int e = 0; e < 3; ++e) {
        int c = t + e * 128;
        vals[e] = s[i * CS + c] + x[i * CS + c];
        sum += vals[e];
    }
    red[t] = sum;
    __syncthreads();
    for (int st = 64; st > 0; st >>= 1) {
        if (t < st) red[t] += red[t + st];
        __syncthreads();
    }
    float mu = red[0] * (1.f / CS);
    __syncthreads();
    float vs = 0.f;
#pragma unroll
    for (int e = 0; e < 3; ++e) { float d = vals[e] - mu; vs += d * d; }
    red[t] = vs;
    __syncthreads();
    for (int st = 64; st > 0; st >>= 1) {
        if (t < st) red[t] += red[t + st];
        __syncthreads();
    }
    float inv = rsqrtf(red[0] * (1.f / CS) + 1e-5f);
#pragma unroll
    for (int e = 0; e < 3; ++e) {
        int c = t + e * 128;
        s[i * CS + c] = (vals[e] - mu) * inv * g[c] + b[c];
    }
}

// ---------------------------------------------------------------------------
// coords[i] += (s[i] @ Wu + bu)[:3]
// ---------------------------------------------------------------------------
__global__ __launch_bounds__(96) void upd_coords_kernel(
    const float* __restrict__ s, const float* __restrict__ Wu,
    const float* __restrict__ bu, float* __restrict__ coords)
{
    const int i = blockIdx.x;
    const int t = threadIdx.x;
    const int c = t >> 5;
    const int lane = t & 31;
    float acc = 0.f;
    for (int kk = lane; kk < CS; kk += 32)
        acc += s[i * CS + kk] * Wu[kk * 6 + c];
#pragma unroll
    for (int off = 16; off > 0; off >>= 1)
        acc += __shfl_down(acc, off, 32);
    if (lane == 0)
        coords[i * 3 + c] += acc + bu[c];
}

// ---------------------------------------------------------------------------
extern "C" void kernel_launch(void* const* d_in, const int* in_sizes, int n_in,
                              void* d_out, int out_size, void* d_ws, size_t ws_size,
                              hipStream_t stream)
{
    const float* s_in = (const float*)d_in[0];
    const float* z    = (const float*)d_in[1];
    const float* Wq   = (const float*)d_in[2];
    const float* bq   = (const float*)d_in[3];
    const float* Wk   = (const float*)d_in[4];
    const float* bk   = (const float*)d_in[5];
    const float* Wv   = (const float*)d_in[6];
    const float* bv   = (const float*)d_in[7];
    const float* Wb   = (const float*)d_in[8];
    const float* bb   = (const float*)d_in[9];
    const float* Wo   = (const float*)d_in[10];
    const float* bo   = (const float*)d_in[11];
    const float* Wt1  = (const float*)d_in[12];
    const float* bt1  = (const float*)d_in[13];
    const float* Wt2  = (const float*)d_in[14];
    const float* bt2  = (const float*)d_in[15];
    const float* Wu   = (const float*)d_in[16];
    const float* bu   = (const float*)d_in[17];
    const float* ln_g = (const float*)d_in[18];
    const float* ln_b = (const float*)d_in[19];

    float* coords = (float*)d_out;   // (1,512,3) output lives in d_out

    char* ws = (char*)d_ws;
    size_t off = 0;
    auto alloc = [&](size_t bytes) -> void* {
        void* p = ws + off;
        off = (off + bytes + 255) & ~(size_t)255;
        return p;
    };
    const long long ZN = (long long)NRES * NRES * CZ;
    __bf16* z_bf  = (__bf16*)alloc((size_t)ZN * sizeof(__bf16));        // 67 MB
    float* qb     = (float*)alloc((size_t)NRES * NH * CH * 4);
    float* kb_    = (float*)alloc((size_t)NRES * NH * CH * 4);
    float* vb     = (float*)alloc((size_t)NRES * NH * CH * 4);
    float* bz     = (float*)alloc((size_t)NRES * NRES * 16 * 4);        // padded N 12->16
    float* wgt    = (float*)alloc((size_t)NRES * NH * NRES * 4);
    float* obuf   = (float*)alloc((size_t)NRES * NH * (CH + CZ) * 4);
    float* xbuf   = (float*)alloc((size_t)NRES * CS * 4);
    float* t1     = (float*)alloc((size_t)NRES * 4 * CS * 4);
    float* s_buf  = (float*)alloc((size_t)NRES * CS * 4);

    cvt_bf16_kernel<<<(int)((ZN + 255) / 256), 256, 0, stream>>>(z, z_bf, (int)ZN);
    init_kernel<<<(NRES * CS + 255) / 256, 256, 0, stream>>>(s_in, s_buf, coords);

    const int HD = NH * CH;        // 192
    const int OC = NH * (CH + CZ); // 1728

    for (int r = 0; r < NREC; ++r) {
        for (int l = 0; l < NL; ++l) {
            const float* Wq_l = Wq + (size_t)l * CS * HD;
            const float* Wk_l = Wk + (size_t)l * CS * HD;
            const float* Wv_l = Wv + (size_t)l * CS * HD;
            const float* Wb_l = Wb + (size_t)l * CZ * NH;
            const float* Wo_l = Wo + (size_t)l * OC * CS;
            const float* W1_l = Wt1 + (size_t)l * CS * 4 * CS;
            const float* W2_l = Wt2 + (size_t)l * 4 * CS * CS;
            const float* Wu_l = Wu + (size_t)l * CS * 6;

            // q / k / v projections: (512x384)@(384x192), tiles 64x64
            gemm_wmma<64, 64, 64, 2, 4, false, false, float, float>
                <<<dim3(HD / 64, NRES / 64, 1), 256, 0, stream>>>(
                s_buf, CS, 0, Wq_l, HD, 0, qb, HD, 0, bq + (size_t)l * HD,
                NRES, HD, CS, 0);
            gemm_wmma<64, 64, 64, 2, 4, false, false, float, float>
                <<<dim3(HD / 64, NRES / 64, 1), 256, 0, stream>>>(
                s_buf, CS, 0, Wk_l, HD, 0, kb_, HD, 0, bk + (size_t)l * HD,
                NRES, HD, CS, 0);
            gemm_wmma<64, 64, 64, 2, 4, false, false, float, float>
                <<<dim3(HD / 64, NRES / 64, 1), 256, 0, stream>>>(
                s_buf, CS, 0, Wv_l, HD, 0, vb, HD, 0, bv + (size_t)l * HD,
                NRES, HD, CS, 0);

            // pair bias: (512*512 x 128)@(128 x 12) -> bz (ldc=16), N-guarded
            gemm_wmma<128, 16, 64, 8, 1, false, true, __bf16, float>
                <<<dim3(1, (NRES * NRES) / 128, 1), 256, 0, stream>>>(
                z_bf, CZ, 0, Wb_l, NH, 0, bz, 16, 0, bb + (size_t)l * NH,
                NRES * NRES, NH, CZ, 0);

            // fused scores + softmax -> wgt[i][h][j]
            attn_softmax_kernel<<<dim3(NRES, NH, 1), 256, 0, stream>>>(
                qb, kb_, bz, coords, wgt);

            // scalar head output -> obuf[:, h*144 + 0..15]
            attn_scalar_kernel<<<dim3(NRES, NH, 1), 64, 0, stream>>>(wgt, vb, obuf);

            // pair contraction (batched per i): (12x512)@(512x128), M-guarded
            gemm_wmma<16, 128, 64, 1, 8, true, false, float, __bf16>
                <<<dim3(1, 1, NRES), 256, 0, stream>>>(
                wgt, NRES, (long long)NH * NRES,
                z_bf, CZ, (long long)NRES * CZ,
                obuf + CH, CH + CZ, (long long)OC,
                nullptr, NH, CZ, NRES, 0);

            // output projection: (512x1728)@(1728x384), tiles 64x128
            gemm_wmma<64, 128, 64, 2, 4, false, false, float, float>
                <<<dim3(CS / 128, NRES / 64, 1), 256, 0, stream>>>(
                obuf, OC, 0, Wo_l, CS, 0, xbuf, CS, 0, bo + (size_t)l * CS,
                NRES, CS, OC, 0);

            add_ln_kernel<<<NRES, 128, 0, stream>>>(
                s_buf, xbuf, ln_g + (size_t)(2 * l) * CS, ln_b + (size_t)(2 * l) * CS);

            // FFN
            gemm_wmma<64, 128, 64, 2, 4, false, false, float, float>
                <<<dim3((4 * CS) / 128, NRES / 64, 1), 256, 0, stream>>>(
                s_buf, CS, 0, W1_l, 4 * CS, 0, t1, 4 * CS, 0, bt1 + (size_t)l * 4 * CS,
                NRES, 4 * CS, CS, 1);
            gemm_wmma<64, 128, 64, 2, 4, false, false, float, float>
                <<<dim3(CS / 128, NRES / 64, 1), 256, 0, stream>>>(
                t1, 4 * CS, 0, W2_l, CS, 0, xbuf, CS, 0, bt2 + (size_t)l * CS,
                NRES, CS, 4 * CS, 0);

            add_ln_kernel<<<NRES, 128, 0, stream>>>(
                s_buf, xbuf, ln_g + (size_t)(2 * l + 1) * CS, ln_b + (size_t)(2 * l + 1) * CS);

            upd_coords_kernel<<<NRES, 96, 0, stream>>>(s_buf, Wu_l, bu + (size_t)l * 6, coords);
        }
    }
}
